// SetEncoderPointNet_86122684220290
// MI455X (gfx1250) — compile-verified, matrix operand order checked by reading, not run on hardware
//
#include <hip/hip_runtime.h>
#include <hip/hip_bf16.h>

typedef __attribute__((ext_vector_type(16))) __bf16 v16bf;
typedef __attribute__((ext_vector_type(8)))  float  v8f;

union Frag { v16bf v; unsigned int u[8]; };

#define CIN   64
#define CMID  256
#define COUT  256
#define MDIM  256
#define NBN   2048            // B*N
#define XSTR  36              // uints per LDS row: 32 (64 bf16) + 4 pad (bank-conflict break)

// f32 pair -> packed bf16 dword; plain casts let clang use the HW packed cvt
__device__ __forceinline__ unsigned int pack_bf16(float lo, float hi) {
    __bf16 a = (__bf16)lo;
    __bf16 b = (__bf16)hi;
    unsigned short ua = __builtin_bit_cast(unsigned short, a);
    unsigned short ub = __builtin_bit_cast(unsigned short, b);
    return (unsigned int)ua | ((unsigned int)ub << 16);
}

// bf16-pair (dword) index inside a 32-wide K slab for A/B fragment slot p.
// ISA 7.12.2: lanes<16 hold K pairs {0,2,4,6,16,18,20,22}; lanes>=16 the +8 set.
__device__ __forceinline__ int frag_kidx(int p, int laneHi) {
    return ((p & 4) << 1) + (p & 3) + (laneHi << 2);
}

// load one B fragment pair (K=64 -> two 32-slabs) from a 64-float weight row
__device__ __forceinline__ void load_b64row(const float* __restrict__ wrow,
                                            int laneHi, Frag& f0, Frag& f1) {
#pragma unroll
    for (int p = 0; p < 8; ++p) {
        int k0 = frag_kidx(p, laneHi);
        float2 a = ((const float2*)wrow)[k0];
        float2 b = ((const float2*)wrow)[k0 + 16];
        f0.u[p] = pack_bf16(a.x, a.y);
        f1.u[p] = pack_bf16(b.x, b.y);
    }
}

// ---------------- Kernel 1: z = x@w1^T (+b1), column-max over M -> zv[bn][256]
__global__ __launch_bounds__(256) void k1_gemm_max(
    const float* __restrict__ x, const float* __restrict__ w1,
    const float* __restrict__ bias, float* __restrict__ zv)
{
    __shared__ unsigned int xlds[MDIM * XSTR];
    const int bn  = blockIdx.x;
    const int tid = threadIdx.x;

    // stage x[bn, tid, :] as bf16
    const float* xrow = x + (size_t)bn * (MDIM * CIN) + (size_t)tid * CIN;
#pragma unroll
    for (int j = 0; j < 16; ++j) {
        float4 f = ((const float4*)xrow)[j];
        *((uint2*)&xlds[tid * XSTR + j * 2]) =
            make_uint2(pack_bf16(f.x, f.y), pack_bf16(f.z, f.w));
    }
    __syncthreads();

    const int lane = tid & 31, wave = tid >> 5;
    const int laneHi = lane >> 4, lane15 = lane & 15;

    // this wave owns two adjacent column tiles; keep all 4 B frags resident
    const int d0 = (wave * 2 + 0) * 16 + lane15;
    const int d1 = (wave * 2 + 1) * 16 + lane15;
    Frag fb00, fb01, fb10, fb11;
    load_b64row(w1 + (size_t)d0 * CIN, laneHi, fb00, fb01);
    load_b64row(w1 + (size_t)d1 * CIN, laneHi, fb10, fb11);

    float vmax0 = -3.402823466e+38f, vmax1 = -3.402823466e+38f;
    for (int mt = 0; mt < 16; ++mt) {
        Frag fa0, fa1;
        const int row = mt * 16 + lane15;
#pragma unroll
        for (int p = 0; p < 8; ++p) {
            int k0 = frag_kidx(p, laneHi);
            fa0.u[p] = xlds[row * XSTR + k0];
            fa1.u[p] = xlds[row * XSTR + k0 + 16];
        }
        v8f c0 = {}, c1 = {};
        c0 = __builtin_amdgcn_wmma_f32_16x16x32_bf16(false, fa0.v, false, fb00.v,
                                                     (short)0, c0, false, false);
        c0 = __builtin_amdgcn_wmma_f32_16x16x32_bf16(false, fa1.v, false, fb01.v,
                                                     (short)0, c0, false, false);
        c1 = __builtin_amdgcn_wmma_f32_16x16x32_bf16(false, fa0.v, false, fb10.v,
                                                     (short)0, c1, false, false);
        c1 = __builtin_amdgcn_wmma_f32_16x16x32_bf16(false, fa1.v, false, fb11.v,
                                                     (short)0, c1, false, false);
#pragma unroll
        for (int r = 0; r < 8; ++r) {
            vmax0 = fmaxf(vmax0, c0[r]);
            vmax1 = fmaxf(vmax1, c1[r]);
        }
    }
    // fold the two lane-halves (same channel, different M rows)
    vmax0 = fmaxf(vmax0, __shfl_xor(vmax0, 16, 32));
    vmax1 = fmaxf(vmax1, __shfl_xor(vmax1, 16, 32));
    if (laneHi == 0) {
        zv[(size_t)bn * CMID + d0] = vmax0 + bias[d0];   // +b1 commutes with max
        zv[(size_t)bn * CMID + d1] = vmax1 + bias[d1];
    }
}

// ---------------- Kernel 2: v = zv @ w2_z^T  (2048x256)(256x256), one wave per tile
__global__ __launch_bounds__(32) void k2_vertex(
    const float* __restrict__ zv, const float* __restrict__ w2,
    float* __restrict__ v)
{
    const int tile = blockIdx.x;              // 128 row blocks x 16 col tiles
    const int rb = tile >> 4, nt = tile & 15;
    const int lane = threadIdx.x;
    const int laneHi = lane >> 4, lane15 = lane & 15;
    const int arow = rb * 16 + lane15;
    const int d    = nt * 16 + lane15;
    const float* zr = zv + (size_t)arow * CMID;
    const float* wr = w2 + (size_t)d * (CIN + CMID) + CIN;   // w2_z part
    v8f c = {};
#pragma unroll
    for (int kb = 0; kb < 8; ++kb) {          // K = 256 in slabs of 32
        Frag fa, fb;
#pragma unroll
        for (int p = 0; p < 8; ++p) {
            int k0 = frag_kidx(p, laneHi) + kb * 16;
            float2 a2 = ((const float2*)zr)[k0];
            float2 b2 = ((const float2*)wr)[k0];
            fa.u[p] = pack_bf16(a2.x, a2.y);
            fb.u[p] = pack_bf16(b2.x, b2.y);
        }
        c = __builtin_amdgcn_wmma_f32_16x16x32_bf16(false, fa.v, false, fb.v,
                                                    (short)0, c, false, false);
    }
#pragma unroll
    for (int r = 0; r < 8; ++r) {
        int row = rb * 16 + r + laneHi * 8;
        v[(size_t)row * COUT + d] = c[r];
    }
}

// ---------------- Kernel 3: out = x @ w2_x^T + broadcast v[bn]
__global__ __launch_bounds__(256) void k3_out(
    const float* __restrict__ x, const float* __restrict__ w2,
    const float* __restrict__ v, float* __restrict__ out)
{
    __shared__ unsigned int xlds[MDIM * XSTR];
    __shared__ float vlds[COUT];
    const int bn  = blockIdx.x;
    const int tid = threadIdx.x;

    const float* xrow = x + (size_t)bn * (MDIM * CIN) + (size_t)tid * CIN;
#pragma unroll
    for (int j = 0; j < 16; ++j) {
        float4 f = ((const float4*)xrow)[j];
        *((uint2*)&xlds[tid * XSTR + j * 2]) =
            make_uint2(pack_bf16(f.x, f.y), pack_bf16(f.z, f.w));
    }
    vlds[tid] = v[(size_t)bn * COUT + tid];
    __syncthreads();

    const int lane = tid & 31, wave = tid >> 5;
    const int laneHi = lane >> 4, lane15 = lane & 15;
    float* outBase = out + (size_t)bn * (MDIM * COUT);

    const int d0 = (wave * 2 + 0) * 16 + lane15;
    const int d1 = (wave * 2 + 1) * 16 + lane15;
    Frag fb00, fb01, fb10, fb11;
    load_b64row(w2 + (size_t)d0 * (CIN + CMID), laneHi, fb00, fb01);  // w2_x part
    load_b64row(w2 + (size_t)d1 * (CIN + CMID), laneHi, fb10, fb11);
    const float vd0 = vlds[d0];   // same for both lane halves (N = lane&15)
    const float vd1 = vlds[d1];

    for (int mt = 0; mt < 16; ++mt) {
        Frag fa0, fa1;
        const int row = mt * 16 + lane15;
#pragma unroll
        for (int p = 0; p < 8; ++p) {
            int k0 = frag_kidx(p, laneHi);
            fa0.u[p] = xlds[row * XSTR + k0];
            fa1.u[p] = xlds[row * XSTR + k0 + 16];
        }
        v8f c0 = {vd0, vd0, vd0, vd0, vd0, vd0, vd0, vd0};  // broadcast-add via C init
        v8f c1 = {vd1, vd1, vd1, vd1, vd1, vd1, vd1, vd1};
        c0 = __builtin_amdgcn_wmma_f32_16x16x32_bf16(false, fa0.v, false, fb00.v,
                                                     (short)0, c0, false, false);
        c0 = __builtin_amdgcn_wmma_f32_16x16x32_bf16(false, fa1.v, false, fb01.v,
                                                     (short)0, c0, false, false);
        c1 = __builtin_amdgcn_wmma_f32_16x16x32_bf16(false, fa0.v, false, fb10.v,
                                                     (short)0, c1, false, false);
        c1 = __builtin_amdgcn_wmma_f32_16x16x32_bf16(false, fa1.v, false, fb11.v,
                                                     (short)0, c1, false, false);
#pragma unroll
        for (int r = 0; r < 8; ++r) {
            int m = mt * 16 + r + laneHi * 8;
            outBase[(size_t)m * COUT + d0] = c0[r];
            outBase[(size_t)m * COUT + d1] = c1[r];
        }
    }
}

extern "C" void kernel_launch(void* const* d_in, const int* in_sizes, int n_in,
                              void* d_out, int out_size, void* d_ws, size_t ws_size,
                              hipStream_t stream) {
    (void)in_sizes; (void)n_in; (void)out_size; (void)ws_size;
    const float* x  = (const float*)d_in[0];   // (8,256,256,64)
    const float* w1 = (const float*)d_in[1];   // (256,64)
    const float* b1 = (const float*)d_in[2];   // (256)
    const float* w2 = (const float*)d_in[3];   // (256,320)
    float* out = (float*)d_out;                // (8,256,256,256)

    float* zv = (float*)d_ws;                  // 2048*256 f32 = 2 MB
    float* v  = zv + (size_t)NBN * CMID;       // 2048*256 f32 = 2 MB

    k1_gemm_max<<<NBN, 256, 0, stream>>>(x, w1, b1, zv);
    k2_vertex  <<<NBN, 32, 0, stream>>>(zv, w2, v);
    k3_out     <<<NBN, 256, 0, stream>>>(x, w2, v, out);
}
